// LaplacianPositionalEncoding_31791347925001
// MI455X (gfx1250) — matrix operand order repaired and checked by reading, not compiled, not run on hardware
//
#include <hip/hip_runtime.h>
#include <hip/hip_bf16.h>

typedef unsigned short u16;
typedef unsigned int   u32;
typedef __attribute__((ext_vector_type(4)))  u32    u32x4;
typedef __attribute__((ext_vector_type(8)))  u32    u32x8;
typedef __attribute__((ext_vector_type(16))) __bf16 v16bf;
typedef __attribute__((ext_vector_type(8)))  float  v8f;

#define NN    4096          // num_nodes
#define PD    128           // POS_DIM
#define ITERS 30
#define KP    128           // K-panel staged in LDS per block
#define LDS_STRIDE 136      // bf16 elems per row: 272 B = 17*16 B -> conflict-light ds_load_b128

__device__ __forceinline__ u16 f2bf(float x) {
    u32 u = __builtin_bit_cast(u32, x);
    u32 r = u + 0x7FFFu + ((u >> 16) & 1u);   // round-to-nearest-even
    return (u16)(r >> 16);
}
__device__ __forceinline__ float bf2f(u16 h) {
    u32 u = ((u32)h) << 16;
    return __builtin_bit_cast(float, u);
}

// ---------------- build adjacency ----------------
__global__ void zero_kernel(u32x4* __restrict__ p, size_t n4) {
    size_t stride = (size_t)gridDim.x * blockDim.x;
    for (size_t i = (size_t)blockIdx.x * blockDim.x + threadIdx.x; i < n4; i += stride)
        p[i] = (u32x4){0u, 0u, 0u, 0u};
}

__global__ void scatter_kernel(const float* __restrict__ w, const int* __restrict__ ei,
                               int E, u16* __restrict__ A) {
    int e = blockIdx.x * blockDim.x + threadIdx.x;
    if (e < E) {
        int i = ei[e], j = ei[E + e];
        u16 v = f2bf(w[e]);                       // .set semantics: races store same value
        A[(size_t)i * NN + j] = v;
        A[(size_t)j * NN + i] = v;
    }
}

__global__ void deg_kernel(const u16* __restrict__ A, float* __restrict__ dinv) {
    __shared__ float sm[256];
    int row = blockIdx.x;
    const u16* Ar = A + (size_t)row * NN;
    float s = 0.f;
    for (int j = threadIdx.x; j < NN; j += 256) s += bf2f(Ar[j]);
    sm[threadIdx.x] = s; __syncthreads();
    for (int off = 128; off; off >>= 1) {
        if (threadIdx.x < off) sm[threadIdx.x] += sm[threadIdx.x + off];
        __syncthreads();
    }
    if (threadIdx.x == 0) dinv[row] = rsqrtf(sm[0] + 1e-5f);   // EPS from reference
}

__global__ void normalize_kernel(u16* __restrict__ A, const float* __restrict__ dinv) {
    size_t total = (size_t)NN * NN;
    size_t stride = (size_t)gridDim.x * blockDim.x;
    for (size_t id = (size_t)blockIdx.x * blockDim.x + threadIdx.x; id < total; id += stride) {
        int m = (int)(id >> 12), j = (int)(id & 4095);
        A[id] = f2bf(bf2f(A[id]) * dinv[m] * dinv[j]);
    }
}

// ---------------- subspace iteration ----------------
__global__ void init_x_kernel(u16* __restrict__ Xt) {
    size_t total = (size_t)PD * NN;
    size_t stride = (size_t)gridDim.x * blockDim.x;
    for (size_t id = (size_t)blockIdx.x * blockDim.x + threadIdx.x; id < total; id += stride) {
        u32 h = (u32)id * 2654435761u;
        h ^= h >> 16; h *= 2246822519u; h ^= h >> 13;
        float v = ((float)(h & 0xFFFFu) / 32768.0f) - 1.0f;
        Xt[id] = f2bf(v);
    }
}

// Y = Anorm * X + X  (M = I + Anorm), bf16 WMMA with f32 accumulation.
// Anorm: NN x NN bf16 row-major.  Xt: PD x NN bf16 (X transposed).  Y: NN x PD f32.
// Block = 8 waves; wave w handles rows [blk*128 + 16w, +16), all 128 cols (8 WMMA tiles).
// B (Xt) K-panels are staged once per block in LDS and served to all 8 waves via ds_load,
// cutting redundant L2 B-traffic 8x so the loop is bound by the A stream (L2-resident).
__global__ __launch_bounds__(256) void gemm_mx_kernel(const u16* __restrict__ A,
                                                      const u16* __restrict__ Xt,
                                                      float* __restrict__ Y) {
    __shared__ u16 ldsB[PD * LDS_STRIDE];        // 34,816 B of the 320 KB WGP LDS

    const int tid  = threadIdx.x;
    const int wave = tid >> 5;
    const int lane = tid & 31;
    const int r    = lane & 15;
    const int hi   = lane >> 4;
    const int rowBase = blockIdx.x * 128 + wave * 16;
    const u16* Arow = A + (size_t)(rowBase + r) * NN;

    v8f acc[8] = {};
    for (int kb = 0; kb < NN; kb += KP) {
        // ---- stage Xt[:, kb..kb+KP) into LDS: 128 rows x KP cols, 16B chunks ----
        __syncthreads();                          // previous panel fully consumed
        for (int c = tid; c < PD * (KP / 8); c += 256) {   // 2048 chunks / 256 thr = 8 each
            int n  = c >> 4;                      // row (KP/8 == 16 chunks per row)
            int cc = c & 15;                      // 16B chunk within row
            u32x4 v = *(const u32x4*)(Xt + (size_t)n * NN + kb + cc * 8);
            *(u32x4*)(ldsB + n * LDS_STRIDE + cc * 8) = v;
        }
        __syncthreads();

        for (int k2 = 0; k2 < KP; k2 += 32) {
            const int k = kb + k2;
            __builtin_prefetch(Arow + k + 1024, 0, 1);   // global_prefetch_b8 on A stream
            // A fragment 16x32 bf16 (ISA layout): lane r holds K {k+8hi..+7} and {k+16+8hi..+7}
            u32x4 a0 = *(const u32x4*)(Arow + k + hi * 8);
            u32x4 a1 = *(const u32x4*)(Arow + k + 16 + hi * 8);
            u32x8 aw;
            aw[0]=a0[0]; aw[1]=a0[1]; aw[2]=a0[2]; aw[3]=a0[3];
            aw[4]=a1[0]; aw[5]=a1[1]; aw[6]=a1[2]; aw[7]=a1[3];
            v16bf af = __builtin_bit_cast(v16bf, aw);
#pragma unroll
            for (int t = 0; t < 8; ++t) {
                // B fragment 32x16: lane = column t*16+r, K {k2+8hi..+7} and {k2+16+8hi..+7} (LDS)
                const u16* bp = ldsB + (t * 16 + r) * LDS_STRIDE + k2 + hi * 8;
                u32x4 b0 = *(const u32x4*)bp;
                u32x4 b1 = *(const u32x4*)(bp + 16);
                u32x8 bw;
                bw[0]=b0[0]; bw[1]=b0[1]; bw[2]=b0[2]; bw[3]=b0[3];
                bw[4]=b1[0]; bw[5]=b1[1]; bw[6]=b1[2]; bw[7]=b1[3];
                v16bf bf = __builtin_bit_cast(v16bf, bw);
                acc[t] = __builtin_amdgcn_wmma_f32_16x16x32_bf16(
                             false, af, false, bf, (short)0, acc[t], false, false);
            }
        }
    }
#pragma unroll
    for (int t = 0; t < 8; ++t) {
        int col = t * 16 + r;
#pragma unroll
        for (int e = 0; e < 8; ++e) {               // C/D layout: VGPR e -> row e + 8*hi
            int row = rowBase + e + 8 * hi;
            Y[(size_t)row * PD + col] = acc[t][e] + bf2f(Xt[(size_t)col * NN + row]);
        }
    }
}

__global__ void gram_kernel(const float* __restrict__ Y, float* __restrict__ G) {
    int i = blockIdx.y * 16 + threadIdx.y;
    int j = blockIdx.x * 16 + threadIdx.x;
    float s = 0.f;
    for (int m = 0; m < NN; ++m) s += Y[(size_t)m * PD + i] * Y[(size_t)m * PD + j];
    G[(size_t)i * PD + j] = s;
}

// In-place Cholesky of G (upper R) then R^{-1} into Rinv. One block of 128 threads.
__global__ __launch_bounds__(128) void cholinv_kernel(float* __restrict__ G, float* __restrict__ Rinv) {
    int tid = threadIdx.x;
    G[(size_t)tid * PD + tid] += 1e-5f * G[(size_t)tid * PD + tid] + 1e-12f;  // ridge
    __syncthreads();
    for (int k = 0; k < PD; ++k) {
        if (tid == 0) G[(size_t)k * PD + k] = sqrtf(fmaxf(G[(size_t)k * PD + k], 1e-20f));
        __syncthreads();
        float rkk = G[(size_t)k * PD + k];
        if (tid > k) G[(size_t)k * PD + tid] /= rkk;
        __syncthreads();
        if (tid > k) {
            float rkj = G[(size_t)k * PD + tid];          // column j = tid
            for (int i = k + 1; i <= tid; ++i)
                G[(size_t)i * PD + tid] -= G[(size_t)k * PD + i] * rkj;
        }
        __syncthreads();
    }
    int j = tid;                                          // back-substitute column j
    Rinv[(size_t)j * PD + j] = 1.0f / G[(size_t)j * PD + j];
    for (int i = j - 1; i >= 0; --i) {
        float s = 0.f;
        for (int k2 = i + 1; k2 <= j; ++k2)
            s += G[(size_t)i * PD + k2] * Rinv[(size_t)k2 * PD + j];
        Rinv[(size_t)i * PD + j] = -s / G[(size_t)i * PD + i];
    }
    for (int i = j + 1; i < PD; ++i) Rinv[(size_t)i * PD + j] = 0.0f;
}

// X = Y * Rinv (Rinv upper-triangular); write X f32 (row-major) and Xt bf16 (transposed).
__global__ __launch_bounds__(128) void update_kernel(const float* __restrict__ Y,
                                                     const float* __restrict__ Rinv,
                                                     float* __restrict__ X,
                                                     u16* __restrict__ Xt) {
    int m = blockIdx.x, n = threadIdx.x;
    const float* Yr = Y + (size_t)m * PD;
    float s = 0.f;
    for (int k = 0; k <= n; ++k) s += Yr[k] * Rinv[(size_t)k * PD + n];
    X[(size_t)m * PD + n] = s;
    Xt[(size_t)n * NN + m] = f2bf(s);
}

// ---------------- ordering & output ----------------
__global__ void rayleigh_kernel(const float* __restrict__ X, const float* __restrict__ Y,
                                float* __restrict__ theta) {
    __shared__ float sm[256];
    int n = blockIdx.x;
    float s = 0.f;
    for (int m = threadIdx.x; m < NN; m += 256)
        s += X[(size_t)m * PD + n] * Y[(size_t)m * PD + n];
    sm[threadIdx.x] = s; __syncthreads();
    for (int off = 128; off; off >>= 1) {
        if (threadIdx.x < off) sm[threadIdx.x] += sm[threadIdx.x + off];
        __syncthreads();
    }
    if (threadIdx.x == 0) theta[n] = sm[0];
}

__global__ __launch_bounds__(128) void sort_kernel(const float* __restrict__ theta,
                                                   int* __restrict__ perm) {
    __shared__ float t[PD];
    int tid = threadIdx.x;
    t[tid] = theta[tid]; __syncthreads();
    float mine = t[tid];
    int rank = 0;                 // descending theta(M) == ascending lambda(L)
    for (int k = 0; k < PD; ++k)
        rank += (t[k] > mine) || (t[k] == mine && k < tid);
    perm[rank] = tid;
}

__global__ void writeout_kernel(const float* __restrict__ X, const int* __restrict__ perm,
                                float* __restrict__ out) {
    int idx = blockIdx.x * blockDim.x + threadIdx.x;
    if (idx < NN * PD) {
        int m = idx >> 7, p = idx & (PD - 1);
        out[idx] = X[(size_t)m * PD + perm[p]];
    }
}

extern "C" void kernel_launch(void* const* d_in, const int* in_sizes, int n_in,
                              void* d_out, int out_size, void* d_ws, size_t ws_size,
                              hipStream_t stream) {
    const float* edge_w  = (const float*)d_in[0];
    const int*   edge_ix = (const int*)d_in[1];     // (2, E) int32 row-major
    (void)n_in; (void)out_size; (void)ws_size;
    const int E = in_sizes[1] / 2;

    char* ws = (char*)d_ws;
    u16*   A     = (u16*)(ws);                                     // 32 MiB bf16 adjacency->Anorm
    float* dinv  = (float*)(ws + (size_t)NN * NN * 2);             // 16 KiB
    u16*   Xt    = (u16*)((char*)dinv + (size_t)NN * 4);           // 1 MiB   X^T bf16 (PD x NN)
    float* Y     = (float*)((char*)Xt + (size_t)PD * NN * 2);      // 2 MiB   (NN x PD)
    float* X     = (float*)((char*)Y + (size_t)NN * PD * 4);       // 2 MiB   (NN x PD)
    float* G     = (float*)((char*)X + (size_t)NN * PD * 4);       // 64 KiB
    float* Rinv  = (float*)((char*)G + (size_t)PD * PD * 4);       // 64 KiB
    float* theta = (float*)((char*)Rinv + (size_t)PD * PD * 4);    // 512 B
    int*   perm  = (int*)((char*)theta + (size_t)PD * 4);          // 512 B

    // Build Anorm = D^{-1/2} A D^{-1/2} in bf16 (fits L2 -> GEMM is WMMA-bound)
    zero_kernel<<<2048, 256, 0, stream>>>((u32x4*)A, (size_t)NN * NN * 2 / 16);
    scatter_kernel<<<(E + 255) / 256, 256, 0, stream>>>(edge_w, edge_ix, E, A);
    deg_kernel<<<NN, 256, 0, stream>>>(A, dinv);
    normalize_kernel<<<4096, 256, 0, stream>>>(A, dinv);

    init_x_kernel<<<1024, 256, 0, stream>>>(Xt);

    dim3 gblk(256), ggrid(NN / 128);
    dim3 qblk(16, 16), qgrid(PD / 16, PD / 16);
    for (int it = 0; it < ITERS; ++it) {
        gemm_mx_kernel<<<ggrid, gblk, 0, stream>>>(A, Xt, Y);     // Y = (I + Anorm) X  [WMMA]
        gram_kernel<<<qgrid, qblk, 0, stream>>>(Y, G);            // G = Y^T Y
        cholinv_kernel<<<1, 128, 0, stream>>>(G, Rinv);           // G = R^T R ; Rinv = R^{-1}
        update_kernel<<<NN, 128, 0, stream>>>(Y, Rinv, X, Xt);    // X = Y Rinv (orthonormal)
    }

    gemm_mx_kernel<<<ggrid, gblk, 0, stream>>>(A, Xt, Y);         // Y = M X for Rayleigh
    rayleigh_kernel<<<PD, 256, 0, stream>>>(X, Y, theta);
    sort_kernel<<<1, 128, 0, stream>>>(theta, perm);
    writeout_kernel<<<(NN * PD + 255) / 256, 256, 0, stream>>>(X, perm, (float*)d_out);
}
